// StateBayesianKalmanNet_39118562132408
// MI455X (gfx1250) — compile-verified, matrix operand order checked by reading, not run on hardware
//
#include <hip/hip_runtime.h>
#include <hip/hip_bf16.h>

// ---------------------------------------------------------------------------
// StateBayesianKalmanNet forward, MI455X (gfx1250, wave32, WMMA).
//
// Roofline: ~71.5 GFLOP of GEMM vs ~284 MB of unavoidable HBM traffic
// (u_fc2 134MB, h_ens 63MB, u_fc1 63MB, ...). 23.3 TB/s -> ~12us memory;
// the GEMMs run as v_wmma_f32_16x16x32_f16 (f16 in, f32 accumulate) with
// weights converted to padded f16 once per launch (<1MB, L2-resident).
// Concrete-dropout / ReLU / full GRU gate math are fused into GEMM
// epilogues so gi/gh (2x180MB) are never materialized and each u_* tensor
// is read exactly once.
//
// Register blocking: plain 1-tile/wave WMMA is only 8 FLOP per L2 byte
// (~9GB of L2 traffic for this net) -> L2-bound. We block 4x2 output tiles
// per wave in the generic GEMM (~21 FLOP/B) and 2 row-tiles in the fused
// GRU (~20 FLOP/B), keeping accumulators at 64/96 VGPRs respectively.
// ---------------------------------------------------------------------------

typedef __attribute__((ext_vector_type(16))) _Float16 v16h;
typedef __attribute__((ext_vector_type(8)))  _Float16 v8h;
typedef __attribute__((ext_vector_type(8)))  float    v8f;

#define S_   32
#define B_   2048
#define M_   16
#define N_   8
#define H_   240
#define HP   256          // H padded to K-multiple of 32 (and 16B rows)
#define IN_  48
#define INP  64           // IN padded
#define O1_  512
#define MN_  128
#define ROWS (S_*B_)      // 65536 (s,b) rows

// ---- device helpers -------------------------------------------------------

__device__ __forceinline__ float sigm(float x) { return 1.f / (1.f + __expf(-x)); }

// concrete dropout: p = sigmoid(p_logit); z = sigmoid((p_logit + log u - log(1-u))/0.1)
__device__ __forceinline__ float cdropf(float x, float u, float plogit, float inv1mp) {
    float z = sigm((plogit + __logf(u) - __logf(1.f - u)) * 10.f);
    return x * (1.f - z) * inv1mp;
}

// Load a 16x16x32 f16 WMMA fragment row for this lane.
// ISA A-matrix 16x32 layout (wave32): lane (m = lane&15, hi = lane>>4),
// element i -> K = (i%8) + hi*8 + (i/8)*16.  That's two contiguous 16B chunks
// at K = hi*8 and K = 16 + hi*8.  B fragment (W rows along K) loads identically.
__device__ __forceinline__ v16h load_frag(const _Float16* __restrict__ row, int k0, int hi) {
    const _Float16* q = row + k0 + hi * 8;
    v8h x0 = *(const v8h*)(q);
    v8h x1 = *(const v8h*)(q + 16);
    return __builtin_shufflevector(x0, x1, 0,1,2,3,4,5,6,7,8,9,10,11,12,13,14,15);
}

__device__ __forceinline__ v8f wmma_f16(v16h a, v16h b, v8f c) {
    return __builtin_amdgcn_wmma_f32_16x16x32_f16(false, a, false, b, (short)0, c, false, false);
}

// ---- feature / innovation build ------------------------------------------
// feat[b,48] = [xf - x_pred_prev_mean, innov, xf - x_prev_prev_mean, y_t - y_prev]
__global__ void k_feat(const float* __restrict__ y_t, const float* __restrict__ x_ens,
                       const float* __restrict__ x_prev_prev, const float* __restrict__ x_pred_prev,
                       const float* __restrict__ y_prev,
                       const float* __restrict__ F, const float* __restrict__ Hm,
                       float* __restrict__ feat, float* __restrict__ innov) {
    int b = blockIdx.x * blockDim.x + threadIdx.x;
    if (b >= B_) return;
    float xf[M_];
    #pragma unroll
    for (int m = 0; m < M_; ++m) xf[m] = 0.f;
    for (int s = 0; s < S_; ++s) {
        const float* p = x_ens + ((size_t)s * B_ + b) * M_;
        #pragma unroll
        for (int m = 0; m < M_; ++m) xf[m] += p[m];
    }
    #pragma unroll
    for (int m = 0; m < M_; ++m) xf[m] *= (1.f / (float)S_);
    float xpm[M_];
    #pragma unroll
    for (int k = 0; k < M_; ++k) {            // x_pred_mean = xf @ F^T  (F[k,m])
        float a = 0.f;
        #pragma unroll
        for (int m = 0; m < M_; ++m) a += xf[m] * F[k * M_ + m];
        xpm[k] = a;
    }
    float inn[N_];
    #pragma unroll
    for (int n = 0; n < N_; ++n) {            // innov = y_t - x_pred_mean @ H^T
        float a = 0.f;
        #pragma unroll
        for (int m = 0; m < M_; ++m) a += xpm[m] * Hm[n * M_ + m];
        inn[n] = y_t[b * N_ + n] - a;
    }
    float* fr = feat + (size_t)b * IN_;
    #pragma unroll
    for (int m = 0; m < M_; ++m) fr[m]      = xf[m] - x_pred_prev[b * M_ + m];
    #pragma unroll
    for (int n = 0; n < N_; ++n) fr[16 + n] = inn[n];
    #pragma unroll
    for (int m = 0; m < M_; ++m) fr[24 + m] = xf[m] - x_prev_prev[b * M_ + m];
    #pragma unroll
    for (int n = 0; n < N_; ++n) fr[40 + n] = y_t[b * N_ + n] - y_prev[b * N_ + n];
    #pragma unroll
    for (int n = 0; n < N_; ++n) innov[b * N_ + n] = inn[n];
}

// ---- f32 -> padded f16 weight conversion ---------------------------------
__global__ void k_convw(const float* __restrict__ src, _Float16* __restrict__ dst,
                        int O, int Kin, int Opad, int Kpad) {
    int idx = blockIdx.x * blockDim.x + threadIdx.x;
    if (idx >= Opad * Kpad) return;
    int o = idx / Kpad, k = idx % Kpad;
    float v = (o < O && k < Kin) ? src[(size_t)o * Kin + k] : 0.f;
    dst[idx] = (_Float16)v;
}

// a_in[sb, 0:64] = cdrop(feat[b,k], u_in[sb,k]) (f16, zero-padded K)
__global__ void k_prep_ain(const float* __restrict__ feat, const float* __restrict__ u_in,
                           const float* __restrict__ p_ptr, _Float16* __restrict__ ain) {
    int idx = blockIdx.x * blockDim.x + threadIdx.x;
    if (idx >= ROWS * INP) return;
    int sb = idx >> 6, k = idx & 63;
    float v = 0.f;
    if (k < IN_) {
        float plogit = *p_ptr;
        float inv1mp = 1.f / (1.f - sigm(plogit));
        int b = sb & (B_ - 1);
        v = cdropf(feat[(size_t)b * IN_ + k], u_in[(size_t)sb * IN_ + k], plogit, inv1mp);
    }
    ain[idx] = (_Float16)v;
}

// h16[sb, 0:256] = (f16)h_ens[sb, 0:240], zero-padded
__global__ void k_prep_h16(const float* __restrict__ h_ens, _Float16* __restrict__ h16) {
    int idx = blockIdx.x * blockDim.x + threadIdx.x;
    if (idx >= ROWS * HP) return;
    int sb = idx >> 8, k = idx & 255;
    float v = (k < H_) ? h_ens[(size_t)sb * H_ + k] : 0.f;
    h16[idx] = (_Float16)v;
}

// zero pad columns 240..255 of a_fc1 (GRU epilogue only writes 0..239)
__global__ void k_padzero(_Float16* __restrict__ afc1) {
    int idx = blockIdx.x * blockDim.x + threadIdx.x;
    if (idx >= ROWS * 16) return;
    int row = idx >> 4, c = H_ + (idx & 15);
    afc1[(size_t)row * HP + c] = (_Float16)0.f;
}

// ---- generic WMMA GEMM, 4x2 register-blocked: out = X @ W^T (+bias, epilogue)
// Each wave owns a 64-row x 32-col output block: per k-step 4 A frags + 2 B
// frags feed 8 WMMAs (~21 FLOP per L2 byte vs 8 for 1-tile waves).
// EPI 0: relu -> f16       (fc_in -> x1)
// EPI 1: relu+cdrop -> f16 (fc1 -> a_fc2, u = u_fc2)
// EPI 2: +bias -> f32      (fc2 -> K_vec)
template<int EPI, int KSTEPS>
__global__ __launch_bounds__(256)
void k_gemm(const _Float16* __restrict__ A, int lda,
            const _Float16* __restrict__ Bw, int ldb,
            const float* __restrict__ bias, int biasN, int colSuper,
            _Float16* __restrict__ outH, float* __restrict__ outF, int ldo,
            const float* __restrict__ u, const float* __restrict__ p_ptr) {
    const int RT = 4, CT = 2;                       // 64x32 block per wave
    int wave = threadIdx.x >> 5, lane = threadIdx.x & 31;
    long t = (long)blockIdx.x * (blockDim.x >> 5) + wave;
    if (t >= (long)(ROWS / (16 * RT)) * colSuper) return;
    int cs = (int)(t % colSuper), rs = (int)(t / colSuper);
    int r0 = rs * 16 * RT, c0 = cs * 16 * CT;
    int m = lane & 15, hi = lane >> 4;
    const _Float16* arow[RT];
    const _Float16* brow[CT];
    #pragma unroll
    for (int i = 0; i < RT; ++i) arow[i] = A + (size_t)(r0 + 16 * i + m) * lda;
    #pragma unroll
    for (int j = 0; j < CT; ++j) brow[j] = Bw + (size_t)(c0 + 16 * j + m) * ldb;
    v8f acc[RT][CT] = {};
    #pragma unroll
    for (int ks = 0; ks < KSTEPS; ++ks) {
        int k0 = ks * 32;
        if (ks + 1 < KSTEPS) __builtin_prefetch(arow[0] + k0 + 32, 0, 1);
        v16h af[RT];
        #pragma unroll
        for (int i = 0; i < RT; ++i) af[i] = load_frag(arow[i], k0, hi);
        #pragma unroll
        for (int j = 0; j < CT; ++j) {
            v16h bf = load_frag(brow[j], k0, hi);
            #pragma unroll
            for (int i = 0; i < RT; ++i) acc[i][j] = wmma_f16(af[i], bf, acc[i][j]);
        }
    }
    float plogit = 0.f, inv1mp = 1.f;
    if (EPI == 1) { plogit = *p_ptr; inv1mp = 1.f / (1.f - sigm(plogit)); }
    #pragma unroll
    for (int j = 0; j < CT; ++j) {
        int col = c0 + 16 * j + (lane & 15);
        float bv = (col < biasN) ? bias[col] : 0.f;   // pad cols: W rows are zero too
        #pragma unroll
        for (int i = 0; i < RT; ++i) {
            #pragma unroll
            for (int r = 0; r < 8; ++r) {             // C/D: vgpr r -> row r + 8*hi
                int row = r0 + 16 * i + r + hi * 8;
                float v = acc[i][j][r] + bv;
                if (EPI == 0) {
                    v = fmaxf(v, 0.f);
                    outH[(size_t)row * ldo + col] = (_Float16)v;
                } else if (EPI == 1) {
                    v = fmaxf(v, 0.f);
                    v = cdropf(v, u[(size_t)row * ldo + col], plogit, inv1mp);
                    outH[(size_t)row * ldo + col] = (_Float16)v;
                } else {
                    outF[(size_t)row * ldo + col] = v;
                }
            }
        }
    }
    (void)u; (void)p_ptr; (void)outH; (void)outF;
}

// ---- fused GRU: gi/gh never hit memory. 2 row-tiles per wave (32x16 out),
// 12 accumulators; each weight fragment feeds 2 WMMAs (~20 FLOP per L2 byte).
__global__ __launch_bounds__(256)
void k_gru(const _Float16* __restrict__ X1, const _Float16* __restrict__ H16,
           const _Float16* __restrict__ Wih, const _Float16* __restrict__ Whh,
           const float* __restrict__ b_ih, const float* __restrict__ b_hh,
           const float* __restrict__ h_ens, const float* __restrict__ u_fc1,
           const float* __restrict__ p_ptr, _Float16* __restrict__ afc1) {
    const int colTiles = H_ / 16;                 // 15
    const int RT = 2;                             // 32-row block per wave
    int wave = threadIdx.x >> 5, lane = threadIdx.x & 31;
    long t = (long)blockIdx.x * (blockDim.x >> 5) + wave;
    if (t >= (long)(ROWS / (16 * RT)) * colTiles) return;
    int ct = (int)(t % colTiles), rs = (int)(t / colTiles);
    int r0 = rs * 16 * RT, c0 = ct * 16;
    int m = lane & 15, hi = lane >> 4;
    const _Float16* ax[RT];
    const _Float16* ah[RT];
    #pragma unroll
    for (int i = 0; i < RT; ++i) {
        ax[i] = X1  + (size_t)(r0 + 16 * i + m) * HP;
        ah[i] = H16 + (size_t)(r0 + 16 * i + m) * HP;
    }
    const _Float16* bg[6];
    #pragma unroll
    for (int g = 0; g < 3; ++g) {
        bg[g]     = Wih + (size_t)(g * H_ + c0 + m) * HP;
        bg[3 + g] = Whh + (size_t)(g * H_ + c0 + m) * HP;
    }
    v8f accI[3][RT] = {};   // i_r, i_z, i_n
    v8f accH[3][RT] = {};   // h_r, h_z, h_n
    #pragma unroll
    for (int ks = 0; ks < HP / 32; ++ks) {        // 8 K-steps, 12 WMMA each
        int k0 = ks * 32;
        v16h axf[RT], ahf[RT];
        #pragma unroll
        for (int i = 0; i < RT; ++i) { axf[i] = load_frag(ax[i], k0, hi); ahf[i] = load_frag(ah[i], k0, hi); }
        #pragma unroll
        for (int g = 0; g < 3; ++g) {
            v16h bf = load_frag(bg[g], k0, hi);
            #pragma unroll
            for (int i = 0; i < RT; ++i) accI[g][i] = wmma_f16(axf[i], bf, accI[g][i]);
        }
        #pragma unroll
        for (int g = 0; g < 3; ++g) {
            v16h bf = load_frag(bg[3 + g], k0, hi);
            #pragma unroll
            for (int i = 0; i < RT; ++i) accH[g][i] = wmma_f16(ahf[i], bf, accH[g][i]);
        }
    }
    float plogit = *p_ptr;
    float inv1mp = 1.f / (1.f - sigm(plogit));
    int col = c0 + (lane & 15);
    float b_ir = b_ih[col], b_iz = b_ih[H_ + col], b_in = b_ih[2 * H_ + col];
    float b_hr = b_hh[col], b_hz = b_hh[H_ + col], b_hn = b_hh[2 * H_ + col];
    #pragma unroll
    for (int i = 0; i < RT; ++i) {
        #pragma unroll
        for (int r = 0; r < 8; ++r) {
            int row = r0 + 16 * i + r + hi * 8;
            float rg = sigm((accI[0][i][r] + b_ir) + (accH[0][i][r] + b_hr));
            float zg = sigm((accI[1][i][r] + b_iz) + (accH[1][i][r] + b_hz));
            float ng = tanhf((accI[2][i][r] + b_in) + rg * (accH[2][i][r] + b_hn));
            float hold = h_ens[(size_t)row * H_ + col];
            float hnew = (1.f - zg) * ng + zg * hold;
            float v = cdropf(hnew, u_fc1[(size_t)row * H_ + col], plogit, inv1mp);
            afc1[(size_t)row * HP + col] = (_Float16)v;
        }
    }
}

// ---- final: x_filt = x_ens @ F^T + K @ innov -----------------------------
__global__ void k_out(const float* __restrict__ x_ens, const float* __restrict__ F,
                      const float* __restrict__ kvec, const float* __restrict__ innov,
                      float* __restrict__ out) {
    int idx = blockIdx.x * blockDim.x + threadIdx.x;
    if (idx >= ROWS * M_) return;
    int sb = idx >> 4, mm = idx & 15, b = sb & (B_ - 1);
    const float* xr = x_ens + (size_t)sb * M_;
    float a = 0.f;
    #pragma unroll
    for (int k = 0; k < M_; ++k) a += xr[k] * F[mm * M_ + k];
    const float* kr = kvec + (size_t)sb * MN_ + mm * N_;
    const float* ir = innov + (size_t)b * N_;
    #pragma unroll
    for (int n = 0; n < N_; ++n) a += kr[n] * ir[n];
    out[idx] = a;
}

// ---- regularizer (scalar, broadcast to [S]) ------------------------------
__global__ void k_reg(const float* __restrict__ W_in, const float* __restrict__ W_fc1,
                      const float* __restrict__ W_fc2,
                      const float* __restrict__ p_in, const float* __restrict__ p_fc1,
                      const float* __restrict__ p_fc2, float* __restrict__ out) {
    __shared__ float s0[256], s1[256], s2[256];
    int t = threadIdx.x;
    float a0 = 0.f, a1 = 0.f, a2 = 0.f;
    for (int i = t; i < H_ * IN_;  i += 256) a0 += W_in[i]  * W_in[i];
    for (int i = t; i < O1_ * H_;  i += 256) a1 += W_fc1[i] * W_fc1[i];
    for (int i = t; i < MN_ * O1_; i += 256) a2 += W_fc2[i] * W_fc2[i];
    s0[t] = a0; s1[t] = a1; s2[t] = a2;
    __syncthreads();
    for (int st = 128; st > 0; st >>= 1) {
        if (t < st) { s0[t] += s0[t + st]; s1[t] += s1[t + st]; s2[t] += s2[t + st]; }
        __syncthreads();
    }
    if (t == 0) {
        auto lr = [](float sumsq, float plog, float fan) {
            float p = 1.f / (1.f + expf(-plog));
            return sumsq / (1.f - p) + fan * (p * logf(p) + (1.f - p) * logf(1.f - p));
        };
        float r = lr(s0[0], *p_in, (float)IN_) + lr(s1[0], *p_fc1, (float)H_)
                + lr(s2[0], *p_fc2, (float)O1_);
        for (int s = 0; s < S_; ++s) out[(size_t)ROWS * M_ + s] = r;
    }
}

// ---------------------------------------------------------------------------
extern "C" void kernel_launch(void* const* d_in, const int* in_sizes, int n_in,
                              void* d_out, int out_size, void* d_ws, size_t ws_size,
                              hipStream_t stream) {
    (void)in_sizes; (void)n_in; (void)out_size; (void)ws_size;
    const float* y_t    = (const float*)d_in[0];
    const float* x_ens  = (const float*)d_in[1];
    const float* xppm   = (const float*)d_in[2];   // x_prev_prev_mean
    const float* xpredp = (const float*)d_in[3];   // x_pred_prev_mean
    const float* y_prev = (const float*)d_in[4];
    const float* h_ens  = (const float*)d_in[5];
    const float* F      = (const float*)d_in[6];
    const float* Hm     = (const float*)d_in[7];
    const float* W_in   = (const float*)d_in[8];
    const float* b_in   = (const float*)d_in[9];
    const float* W_ih   = (const float*)d_in[10];
    const float* b_ih   = (const float*)d_in[11];
    const float* W_hh   = (const float*)d_in[12];
    const float* b_hh   = (const float*)d_in[13];
    const float* W_fc1  = (const float*)d_in[14];
    const float* b_fc1  = (const float*)d_in[15];
    const float* W_fc2  = (const float*)d_in[16];
    const float* b_fc2  = (const float*)d_in[17];
    const float* p_in   = (const float*)d_in[18];
    const float* p_fc1  = (const float*)d_in[19];
    const float* p_fc2  = (const float*)d_in[20];
    const float* u_in   = (const float*)d_in[21];
    const float* u_fc1  = (const float*)d_in[22];
    const float* u_fc2  = (const float*)d_in[23];
    float* out = (float*)d_out;

    // workspace layout (~202 MB, 256B aligned)
    char* ws = (char*)d_ws;
    size_t off = 0;
    auto alloc = [&](size_t bytes) { off = (off + 255) & ~(size_t)255; size_t o = off; off += bytes; return o; };
    float*     feat   = (float*)    (ws + alloc((size_t)B_ * IN_ * 4));
    float*     innov  = (float*)    (ws + alloc((size_t)B_ * N_ * 4));
    _Float16*  Win16  = (_Float16*) (ws + alloc((size_t)HP * INP * 2));       // 256x64, zero-padded
    _Float16*  Wih16  = (_Float16*) (ws + alloc((size_t)3 * H_ * HP * 2));    // 720x256
    _Float16*  Whh16  = (_Float16*) (ws + alloc((size_t)3 * H_ * HP * 2));
    _Float16*  Wfc116 = (_Float16*) (ws + alloc((size_t)O1_ * HP * 2));       // 512x256
    _Float16*  Wfc216 = (_Float16*) (ws + alloc((size_t)MN_ * O1_ * 2));      // 128x512
    _Float16*  ain    = (_Float16*) (ws + alloc((size_t)ROWS * INP * 2));
    _Float16*  x1h    = (_Float16*) (ws + alloc((size_t)ROWS * HP * 2));
    _Float16*  h16    = (_Float16*) (ws + alloc((size_t)ROWS * HP * 2));
    _Float16*  afc1   = (_Float16*) (ws + alloc((size_t)ROWS * HP * 2));
    _Float16*  afc2   = (_Float16*) (ws + alloc((size_t)ROWS * O1_ * 2));
    float*     kvec   = (float*)    (ws + alloc((size_t)ROWS * MN_ * 4));

    const int TB = 256;
    // 1. feature / innovation build
    k_feat<<<B_ / TB, TB, 0, stream>>>(y_t, x_ens, xppm, xpredp, y_prev, F, Hm, feat, innov);
    // 2. weight f16 conversion (zero-padded K / O)
    k_convw<<<(HP * INP + TB - 1) / TB, TB, 0, stream>>>(W_in,  Win16,  H_, IN_, HP, INP);
    k_convw<<<(3 * H_ * HP + TB - 1) / TB, TB, 0, stream>>>(W_ih, Wih16, 3 * H_, H_, 3 * H_, HP);
    k_convw<<<(3 * H_ * HP + TB - 1) / TB, TB, 0, stream>>>(W_hh, Whh16, 3 * H_, H_, 3 * H_, HP);
    k_convw<<<(O1_ * HP + TB - 1) / TB, TB, 0, stream>>>(W_fc1, Wfc116, O1_, H_, O1_, HP);
    k_convw<<<(MN_ * O1_ + TB - 1) / TB, TB, 0, stream>>>(W_fc2, Wfc216, MN_, O1_, MN_, O1_);
    // 3. activation prep (cdrop on feat; f16 h_ens; zero pad cols of a_fc1)
    k_prep_ain<<<(ROWS * INP) / TB, TB, 0, stream>>>(feat, u_in, p_in, ain);
    k_prep_h16<<<(ROWS * HP) / TB, TB, 0, stream>>>(h_ens, h16);
    k_padzero<<<(ROWS * 16) / TB, TB, 0, stream>>>(afc1);
    // 4. GEMM1: x1 = relu(a_in @ W_in^T + b_in)  -> f16 [ROWS, 256]
    k_gemm<0, INP / 32><<<(ROWS / 64) * (HP / 32) / 8, TB, 0, stream>>>(
        ain, INP, Win16, INP, b_in, H_, HP / 32, x1h, nullptr, HP, nullptr, nullptr);
    // 5. fused GRU (gi+gh+gates+cdrop) -> a_fc1 f16 [ROWS, 256]
    k_gru<<<(ROWS / 32) * (H_ / 16) / 8, TB, 0, stream>>>(
        x1h, h16, Wih16, Whh16, b_ih, b_hh, h_ens, u_fc1, p_fc1, afc1);
    // 6. GEMM4: a_fc2 = cdrop(relu(a_fc1 @ W_fc1^T + b_fc1), u_fc2) -> f16 [ROWS, 512]
    k_gemm<1, HP / 32><<<(ROWS / 64) * (O1_ / 32) / 8, TB, 0, stream>>>(
        afc1, HP, Wfc116, HP, b_fc1, O1_, O1_ / 32, afc2, nullptr, O1_, u_fc2, p_fc2);
    // 7. GEMM5: K_vec = a_fc2 @ W_fc2^T + b_fc2 -> f32 [ROWS, 128]
    k_gemm<2, O1_ / 32><<<(ROWS / 64) * (MN_ / 32) / 8, TB, 0, stream>>>(
        afc2, O1_, Wfc216, O1_, b_fc2, MN_, MN_ / 32, nullptr, kvec, MN_, nullptr, nullptr);
    // 8. x_filt = x_pred_ens + K @ innov
    k_out<<<(ROWS * M_) / TB, TB, 0, stream>>>(x_ens, F, kvec, innov, out);
    // 9. regularizer -> out[S*B*M .. +32]
    k_reg<<<1, TB, 0, stream>>>(W_in, W_fc1, W_fc2, p_in, p_fc1, p_fc2, out);
}
